// GraphAttentionLayer_14577119003553
// MI455X (gfx1250) — matrix operand order repaired and checked by reading, not compiled, not run on hardware
//
#include <hip/hip_runtime.h>
#include <hip/hip_bf16.h>

typedef __attribute__((ext_vector_type(16))) _Float16 v16h;
typedef __attribute__((ext_vector_type(8)))  float    v8f;

#define GAT_ALPHA 0.2f
#define GAT_N   4096
#define GAT_IN  256
#define GAT_H   8
#define GAT_D   32

// ---------------------------------------------------------------------------
// Kernel 1: Wh = h @ W[head]  (fp32), emit:
//   WhT  [H][32][N]  f16   (transposed so kernel2 B-operand loads are contiguous)
//   Wh1  [H][N]      f32   (Wh . a1)
//   Wh2  [H][N]      f32   (Wh . a2)
// Block: 256 threads = 8 waves; each wave computes one row n, lane = d.
// ---------------------------------------------------------------------------
__global__ void gat_wh_kernel(const float* __restrict__ h,
                              const float* __restrict__ W,
                              const float* __restrict__ a,
                              _Float16*    __restrict__ WhT,
                              float*       __restrict__ Wh1,
                              float*       __restrict__ Wh2) {
    const int head = blockIdx.y;
    const int n0   = blockIdx.x * 8;

    __shared__ __align__(16) float Wlds[GAT_IN * GAT_D]; // 32KB
    const float* Wg = W + (size_t)head * GAT_IN * GAT_D;
    for (int t = threadIdx.x; t < (GAT_IN * GAT_D) / 4; t += 256)
        ((float4*)Wlds)[t] = ((const float4*)Wg)[t];
    __syncthreads();

    const int lane = threadIdx.x & 31;
    const int wv   = threadIdx.x >> 5;   // 0..7
    const int n    = n0 + wv;
    const int d    = lane;

    const float* hrow = h + (size_t)n * GAT_IN;
    float acc = 0.0f;
#pragma unroll 8
    for (int i = 0; i < GAT_IN; ++i)
        acc = fmaf(hrow[i], Wlds[i * GAT_D + d], acc);

    // transposed f16 copy for kernel2's B operand
    WhT[((size_t)head * GAT_D + d) * GAT_N + n] = (_Float16)acc;

    // attention projection scalars (wave32 reduction over d = lanes)
    const float a1 = a[head * 2 * GAT_D + d];
    const float a2 = a[head * 2 * GAT_D + GAT_D + d];
    float w1 = acc * a1;
    float w2 = acc * a2;
#pragma unroll
    for (int off = 16; off > 0; off >>= 1) {
        w1 += __shfl_xor(w1, off, 32);
        w2 += __shfl_xor(w2, off, 32);
    }
    if (lane == 0) {
        Wh1[head * GAT_N + n] = w1;
        Wh2[head * GAT_N + n] = w2;
    }
}

// ---------------------------------------------------------------------------
// Kernel 2: fused masked-softmax(rank-1 scores) @ Wh via WMMA f16.
// One wave handles a 16-row i-tile; 4 waves per block; Wh2[head] in LDS.
//
// A-operand (P, 16x32 f16) register layout (contiguous k, same as B):
//   lane L: row M = L%16 ;  lo-lanes: K = j0+0..15
//                           hi-lanes: K = j0+16..31
// B-operand (Wh, 32x16 f16): lane L: N(=d) = L%16,
//   lo-lanes K = j0..j0+15, hi-lanes K = j0+16..31 -> one contiguous v16h
//   load from WhT[head][d][j].
// C/D (16x16 f32): lane L: N = L%16, VGPR v: M = v + 8*(L>=16).
// ---------------------------------------------------------------------------
#define K2_WAVES 4

__global__ void gat_attn_kernel(const int*       __restrict__ adj,
                                const _Float16*  __restrict__ WhT,
                                const float*     __restrict__ Wh1,
                                const float*     __restrict__ Wh2,
                                float*           __restrict__ out) {
    const int head  = blockIdx.y;
    const int wave  = threadIdx.x >> 5;
    const int lane  = threadIdx.x & 31;
    const int lhalf = lane >> 4;          // 0: lo K-half, 1: hi K-half
    const int lrow  = lane & 15;          // row within i-tile / N within d-tile
    const int i0    = (blockIdx.x * K2_WAVES + wave) * 16;

    // stage Wh2[head][0..N) into LDS (16KB)
    __shared__ __align__(16) float wh2s[GAT_N];
    {
        const float* g = Wh2 + (size_t)head * GAT_N;
        for (int t = threadIdx.x; t < GAT_N / 4; t += K2_WAVES * 32)
            ((float4*)wh2s)[t] = ((const float4*)g)[t];
    }
    __syncthreads();

    const int    row    = i0 + lrow;
    const float  wh1    = Wh1[head * GAT_N + row];
    const int*   adjrow = adj + (size_t)row * GAT_N;
    const _Float16* b0p = WhT + ((size_t)head * GAT_D + lrow) * GAT_N;
    const _Float16* b1p = b0p + (size_t)16 * GAT_N;

    float m = -1e30f;   // running row max (for row lane%16)
    float s = 0.0f;     // running row denom
    v8f acc0 = {};      // d = 0..15
    v8f acc1 = {};      // d = 16..31

    for (int j0 = 0; j0 < GAT_N; j0 += 32) {
        if (j0 + 128 < GAT_N)
            __builtin_prefetch(adjrow + j0 + 128, 0, 1);   // global_prefetch_b8

        const int kb = j0 + lhalf * 16;

        int avi[16];
        ((int4*)avi)[0] = *(const int4*)(adjrow + kb);
        ((int4*)avi)[1] = *(const int4*)(adjrow + kb + 4);
        ((int4*)avi)[2] = *(const int4*)(adjrow + kb + 8);
        ((int4*)avi)[3] = *(const int4*)(adjrow + kb + 12);

        float wvv[16];
        ((float4*)wvv)[0] = *(const float4*)(wh2s + kb);
        ((float4*)wvv)[1] = *(const float4*)(wh2s + kb + 4);
        ((float4*)wvv)[2] = *(const float4*)(wh2s + kb + 8);
        ((float4*)wvv)[3] = *(const float4*)(wh2s + kb + 12);

        // B operand tiles: contiguous 16 f16 per lane
        const v16h b0 = *(const v16h*)(b0p + j0 + lhalf * 16);
        const v16h b1 = *(const v16h*)(b1p + j0 + lhalf * 16);

        // scores (masked leaky-relu), tile max (tree-reduced)
        float ev[16];
#pragma unroll
        for (int t = 0; t < 16; ++t) {
            float e = wh1 + wvv[t];
            e = (e > 0.0f) ? e : (GAT_ALPHA * e);
            ev[t] = (avi[t] > 0) ? e : -1e30f;
        }
        float mx[8];
#pragma unroll
        for (int t = 0; t < 8; ++t) mx[t] = fmaxf(ev[t], ev[t + 8]);
#pragma unroll
        for (int t = 0; t < 4; ++t) mx[t] = fmaxf(mx[t], mx[t + 4]);
        float tm = fmaxf(fmaxf(mx[0], mx[1]), fmaxf(mx[2], mx[3]));
        tm = fmaxf(tm, __shfl_xor(tm, 16, 32));  // combine lane pair -> row max

        // wave-uniform: does any row see a new max this tile?
        const bool upd = (__ballot(tm > m) != 0ull);

        float mn = m, cf = 1.0f;
        if (upd) {
            mn = fmaxf(m, tm);
            cf = __expf(m - mn);   // accumulator correction for this lane's row
        }

        // probabilities in A layout; masked entries exactly 0
        float pv[16];
        v16h pa;
#pragma unroll
        for (int t = 0; t < 16; ++t) {
            float p = (avi[t] > 0) ? __expf(ev[t] - mn) : 0.0f;
            pv[t] = p;
            pa[t] = (_Float16)p;
        }
        float sm[8];
#pragma unroll
        for (int t = 0; t < 8; ++t) sm[t] = pv[t] + pv[t + 8];
#pragma unroll
        for (int t = 0; t < 4; ++t) sm[t] = sm[t] + sm[t + 4];
        float ts = (sm[0] + sm[1]) + (sm[2] + sm[3]);
        ts += __shfl_xor(ts, 16, 32);

        if (upd) {
            s = s * cf + ts;
            m = mn;
            // rescale accumulators: row of VGPR v is M = v + 8*lhalf;
            // cf for row r lives in lanes r and r+16.
#pragma unroll
            for (int v = 0; v < 8; ++v) {
                const float cv = __shfl(cf, v + 8 * lhalf, 32);
                acc0[v] *= cv;
                acc1[v] *= cv;
            }
        } else {
            s += ts;
        }

        acc0 = __builtin_amdgcn_wmma_f32_16x16x32_f16(
                   false, pa, false, b0, (short)0, acc0, false, false);
        acc1 = __builtin_amdgcn_wmma_f32_16x16x32_f16(
                   false, pa, false, b1, (short)0, acc1, false, false);
    }

    // normalize and store: out[(i0+M), head*32 + d], d = lrow (+16 for acc1)
    const float rinv = __builtin_amdgcn_rcpf(s);   // per-row 1/sum (1-ulp rcp)
#pragma unroll
    for (int v = 0; v < 8; ++v) {
        const float r  = __shfl(rinv, v + 8 * lhalf, 32);
        const int   ri = i0 + v + 8 * lhalf;
        float* orow = out + (size_t)ri * (GAT_H * GAT_D) + head * GAT_D;
        orow[lrow]      = acc0[v] * r;
        orow[16 + lrow] = acc1[v] * r;
    }
}

// ---------------------------------------------------------------------------
extern "C" void kernel_launch(void* const* d_in, const int* in_sizes, int n_in,
                              void* d_out, int out_size, void* d_ws, size_t ws_size,
                              hipStream_t stream) {
    const float* h   = (const float*)d_in[0];   // (4096, 256) f32
    const int*   adj = (const int*)d_in[1];     // (4096, 4096) i32
    const float* W   = (const float*)d_in[2];   // (8, 256, 32) f32
    const float* a   = (const float*)d_in[3];   // (8, 64, 1) f32
    float*       out = (float*)d_out;           // (4096, 256) f32

    char* ws = (char*)d_ws;
    _Float16* WhT = (_Float16*)ws;                                  // 2 MB
    float*    Wh1 = (float*)(ws + 2u * 1024u * 1024u);              // 128 KB
    float*    Wh2 = (float*)(ws + 2u * 1024u * 1024u + 128u * 1024u);

    {
        dim3 grid(GAT_N / 8, GAT_H);
        dim3 block(256);
        gat_wh_kernel<<<grid, block, 0, stream>>>(h, W, a, WhT, Wh1, Wh2);
    }
    {
        dim3 grid(GAT_N / (K2_WAVES * 16), GAT_H);
        dim3 block(K2_WAVES * 32);
        gat_attn_kernel<<<grid, block, 0, stream>>>(adj, WhT, Wh1, Wh2, out);
    }
}
